// Attention_10239202033764
// MI455X (gfx1250) — matrix operand order repaired and checked by reading, not compiled, not run on hardware
//
#include <hip/hip_runtime.h>
#include <hip/hip_bf16.h>

// ---------------------------------------------------------------------------
// Attention (single head, d=512) for MI455X / gfx1250, wave32 + WMMA bf16.
//   out = softmax( (qWq^T)(kWk^T)^T * 64  masked ) @ (vWv^T)
//   1) qkv_proj  : fp32 GEMM -> bf16 qh/kh (row major), vt (transposed V);
//                  register->LDS ping-pong double buffer, 128-VGPR accum/wave
//   2) flash_attn: streaming-softmax attention, async-to-LDS double-buffered
//                  K/V tiles, partial-score split across waves (no dup WMMA)
// WMMA operand streams use interleaved 2-buffer pipelining (both fragments
// simultaneously live -> allocator cannot coalesce them -> counted dscnt
// waits instead of full drains before each WMMA).
// ---------------------------------------------------------------------------

typedef __bf16 bf16;
typedef __attribute__((ext_vector_type(16))) __bf16 v16bf;
typedef __attribute__((ext_vector_type(8)))  __bf16 v8bf;
typedef __attribute__((ext_vector_type(4)))  __bf16 v4bf;
typedef __attribute__((ext_vector_type(8)))  float  v8f;
typedef int v4i_gcc __attribute__((vector_size(4 * sizeof(int))));

#define HIDDEN 4096
#define HEADD  512
#define SEQ    2048
#define NBATCH 4
#define ROWS   (NBATCH * SEQ)      // 8192
#define SCALE_F 64.0f              // sqrt(4096): reference MULTIPLIES by it
#define NEG_F  (-1.0e9f)

// ---- CDNA5 async global->LDS copy (ASYNCcnt) with portable fallback -------
#if defined(__has_builtin)
# if __has_builtin(__builtin_amdgcn_global_load_async_to_lds_b128) && \
     __has_builtin(__builtin_amdgcn_s_wait_asynccnt)
#  define HAS_ASYNC 1
# endif
#endif
#ifndef HAS_ASYNC
# define HAS_ASYNC 0
#endif

__device__ __forceinline__ void cp16(bf16* l, const bf16* g) {
#if HAS_ASYNC
  // ROCm clang signature: generic pointers to 16-byte int vectors
  __builtin_amdgcn_global_load_async_to_lds_b128(
      (v4i_gcc*)g, (v4i_gcc*)l, 0, 0);
#else
  *(uint4*)l = *(const uint4*)g;
#endif
}
__device__ __forceinline__ void async_wait_all() {
#if HAS_ASYNC
  __builtin_amdgcn_s_wait_asynccnt(0);
#endif
}

// Load a 16-element bf16 WMMA operand fragment for this lane from an LDS row.
//   A operand : k0 = half*16, k1 = half*16 + 8   (half = lane>>4)
//   B operand : k0 = half*16, k1 = half*16 + 8
__device__ __forceinline__ v16bf ld_frag(const bf16* p, int k0, int k1) {
  v8bf lo = *(const v8bf*)(p + k0);
  v8bf hi = *(const v8bf*)(p + k1);
  v16bf r;
#pragma unroll
  for (int i = 0; i < 8; ++i) { r[i] = lo[i]; r[8 + i] = hi[i]; }
  return r;
}

__device__ __forceinline__ v8f wmma_bf16(v16bf a, v16bf b, v8f c) {
  return __builtin_amdgcn_wmma_f32_16x16x32_bf16(
      false, a, false, b, (short)0, c, false, false);
}

// ---------------------------------------------------------------------------
// Kernel 1: fused QKV projection.  Grid (ROWS/128, 3), block 512 (16 waves).
// Block tile 128x512: q/k/v each read ONCE from HBM (W is L2-resident).
// Wave tile 16x256 -> 128 accum VGPRs.  K-slab 32, reg->LDS double buffer.
// ---------------------------------------------------------------------------
__global__ __launch_bounds__(512, 1)
void qkv_proj_kernel(const float* __restrict__ q, const float* __restrict__ k,
                     const float* __restrict__ v,
                     const float* __restrict__ Wq, const float* __restrict__ Wk,
                     const float* __restrict__ Wv,
                     bf16* __restrict__ qh, bf16* __restrict__ kh,
                     bf16* __restrict__ vt) {
  const int z = blockIdx.y;                      // 0=q,1=k,2=v
  const float* __restrict__ X = (z == 0) ? q : (z == 1) ? k : v;
  const float* __restrict__ W = (z == 0) ? Wq : (z == 1) ? Wk : Wv;

  // row stride 40 bf16 = 80B (16B aligned, bank-spreading pad)
  __shared__ __align__(16) bf16 As[2][128][40];  // 20.5 KB
  __shared__ __align__(16) bf16 Bs[2][512][40];  // 81.9 KB

  const int tid  = threadIdx.x;
  const int lane = tid & 31, wave = tid >> 5;    // 16 waves
  const int half = lane >> 4, ln = lane & 15;
  const int mbase = blockIdx.x * 128;
  const int wm = (wave & 7) * 16;                // 8 row slices of 16
  const int wn = (wave >> 3) * 256;              // 2 col halves of 256

  float4 fa[2], fb[8];

  // ---- prologue: load + convert + store K-slab 0 into buffer 0
#pragma unroll
  for (int i = 0; i < 2; ++i) {
    int idx = tid + i * 512, r = idx >> 3, c4 = (idx & 7) * 4;
    fa[i] = *(const float4*)(X + (size_t)(mbase + r) * HIDDEN + c4);
  }
#pragma unroll
  for (int i = 0; i < 8; ++i) {
    int idx = tid + i * 512, r = idx >> 3, c4 = (idx & 7) * 4;
    fb[i] = *(const float4*)(W + (size_t)r * HIDDEN + c4);
  }
#pragma unroll
  for (int i = 0; i < 2; ++i) {
    int idx = tid + i * 512, r = idx >> 3, c4 = (idx & 7) * 4;
    v4bf bb; bb[0]=(bf16)fa[i].x; bb[1]=(bf16)fa[i].y; bb[2]=(bf16)fa[i].z; bb[3]=(bf16)fa[i].w;
    *(v4bf*)(&As[0][r][c4]) = bb;
  }
#pragma unroll
  for (int i = 0; i < 8; ++i) {
    int idx = tid + i * 512, r = idx >> 3, c4 = (idx & 7) * 4;
    v4bf bb; bb[0]=(bf16)fb[i].x; bb[1]=(bf16)fb[i].y; bb[2]=(bf16)fb[i].z; bb[3]=(bf16)fb[i].w;
    *(v4bf*)(&Bs[0][r][c4]) = bb;
  }
  __syncthreads();

  v8f acc[16] = {};                              // 16x256 f32 accumulator
  int cur = 0;
#pragma unroll 1
  for (int k0 = 0; k0 < HIDDEN; k0 += 32) {
    const bool more = (k0 + 32) < HIDDEN;
    if (more) {                                  // issue next slab's loads
#pragma unroll
      for (int i = 0; i < 2; ++i) {
        int idx = tid + i * 512, r = idx >> 3, c4 = (idx & 7) * 4;
        fa[i] = *(const float4*)(X + (size_t)(mbase + r) * HIDDEN + k0 + 32 + c4);
      }
#pragma unroll
      for (int i = 0; i < 8; ++i) {
        int idx = tid + i * 512, r = idx >> 3, c4 = (idx & 7) * 4;
        fb[i] = *(const float4*)(W + (size_t)r * HIDDEN + k0 + 32 + c4);
      }
    }
    // compute current slab: 16 WMMAs, interleaved 2-deep B pipeline
    {
      const bf16* brow = &Bs[cur][wn + ln][0];
      v16bf a  = ld_frag(&As[cur][wm + ln][0], half * 16, half * 16 + 8);
      v16bf p0 = ld_frag(brow, half * 16, half * 16 + 8);           // n=0
      v16bf p1 = ld_frag(brow + 16 * 40, half * 16, half * 16 + 8); // n=1
#pragma unroll
      for (int n = 0; n < 16; n += 2) {
        acc[n] = wmma_bf16(a, p0, acc[n]);
        if (n + 2 < 16)
          p0 = ld_frag(brow + (n + 2) * 16 * 40, half * 16, half * 16 + 8);
        acc[n + 1] = wmma_bf16(a, p1, acc[n + 1]);
        if (n + 3 < 16)
          p1 = ld_frag(brow + (n + 3) * 16 * 40, half * 16, half * 16 + 8);
      }
    }
    if (more) {                                  // convert + store next slab
#pragma unroll
      for (int i = 0; i < 2; ++i) {
        int idx = tid + i * 512, r = idx >> 3, c4 = (idx & 7) * 4;
        v4bf bb; bb[0]=(bf16)fa[i].x; bb[1]=(bf16)fa[i].y; bb[2]=(bf16)fa[i].z; bb[3]=(bf16)fa[i].w;
        *(v4bf*)(&As[cur ^ 1][r][c4]) = bb;
      }
#pragma unroll
      for (int i = 0; i < 8; ++i) {
        int idx = tid + i * 512, r = idx >> 3, c4 = (idx & 7) * 4;
        v4bf bb; bb[0]=(bf16)fb[i].x; bb[1]=(bf16)fb[i].y; bb[2]=(bf16)fb[i].z; bb[3]=(bf16)fb[i].w;
        *(v4bf*)(&Bs[cur ^ 1][r][c4]) = bb;
      }
    }
    __syncthreads();
    cur ^= 1;
  }

  // Epilogue.  D layout: lane -> col (ln), VGPR r -> row (half*8 + r).
#pragma unroll
  for (int n = 0; n < 16; ++n)
#pragma unroll
    for (int r = 0; r < 8; ++r) {
      int row = mbase + wm + half * 8 + r;       // global token row
      int col = wn + n * 16 + ln;                // head dim 0..511
      bf16 val = (bf16)acc[n][r];
      if (z == 0)      qh[(size_t)row * HEADD + col] = val;
      else if (z == 1) kh[(size_t)row * HEADD + col] = val;
      else {           // V stored transposed: vt[b][d][s], B-operand-ready
        int bb = row >> 11, s = row & (SEQ - 1);
        vt[((size_t)bb * HEADD + col) * SEQ + s] = val;
      }
    }
}

// ---------------------------------------------------------------------------
// Kernel 2: flash attention.  Grid (SEQ/32, NBATCH) = 256 blocks, 8 waves.
// Block: 32 queries x full d=512.  Waves: 2 query slices x 4 d-quarters.
// Scores split across the 4 d-waves (each does its K-quarter), reduced via
// LDS -> no duplicated WMMA.  K/V tiles double-buffered and prefetched with
// GLOBAL_LOAD_ASYNC_TO_LDS_B128 (ASYNCcnt).
// ---------------------------------------------------------------------------
__global__ __launch_bounds__(256, 1)
void flash_attn_kernel(const bf16* __restrict__ qh, const bf16* __restrict__ kh,
                       const bf16* __restrict__ vt, const int* __restrict__ mask,
                       float* __restrict__ out) {
  __shared__ __align__(16) bf16 Qs[32][520];       // 33.3 KB
  __shared__ __align__(16) bf16 Ks[2][32][520];    // 66.6 KB (double buffer)
  __shared__ __align__(16) bf16 Vs[2][512][40];    // 81.9 KB (double buffer)
  __shared__ __align__(16) bf16 Ps[8][16][40];     // 10.2 KB per-wave P
  __shared__ __align__(16) float Sp[8][2][16][17]; // 17.4 KB score partials
  __shared__ __align__(16) int   Ms[SEQ];          //  8.2 KB mask row

  const int tid  = threadIdx.x;
  const int lane = tid & 31, wave = tid >> 5;
  const int half = lane >> 4, ln = lane & 15;
  const int wq = wave >> 2;                      // query slice 0..1 (16 rows)
  const int wd = wave & 3;                       // d quarter 0..3 (128 cols)
  const int b = blockIdx.y;
  const int qbase = blockIdx.x * 32;

  const bf16* __restrict__ qrow = qh + ((size_t)b * SEQ + qbase) * HEADD;
  const bf16* __restrict__ krow = kh + (size_t)b * SEQ * HEADD;
  const bf16* __restrict__ vrow = vt + (size_t)b * HEADD * SEQ;
  const int*  __restrict__ mrow = mask + b * SEQ;

  // Q tile (32 KB) + mask row, staged once
#pragma unroll
  for (int i = 0; i < 8; ++i) {
    int idx = tid + i * 256;                     // 32 rows x 64 16B-chunks
    int r = idx >> 6, c = (idx & 63) * 8;
    *(uint4*)(&Qs[r][c]) = *(const uint4*)(qrow + (size_t)r * HEADD + c);
    Ms[idx] = mrow[idx];
  }
  // prefetch key/value tile 0 into buffer 0
#pragma unroll
  for (int i = 0; i < 8; ++i) {
    int idx = tid + i * 256;
    int r = idx >> 6, c = (idx & 63) * 8;
    cp16(&Ks[0][r][c], krow + (size_t)r * HEADD + c);
    int d = idx >> 2, c2 = (idx & 3) * 8;
    cp16(&Vs[0][d][c2], vrow + (size_t)d * SEQ + c2);
  }
  async_wait_all();
  __syncthreads();

  v8f acc[8] = {};                               // 16 rows x 128 cols f32
  float mx[8], sm[8];
#pragma unroll
  for (int r = 0; r < 8; ++r) { mx[r] = -3.0e38f; sm[r] = 0.0f; }

  int cur = 0;
#pragma unroll 1
  for (int kb = 0; kb < SEQ; kb += 32) {
    if (kb + 32 < SEQ) {                         // async prefetch next tile
#pragma unroll
      for (int i = 0; i < 8; ++i) {
        int idx = tid + i * 256;
        int r = idx >> 6, c = (idx & 63) * 8;
        cp16(&Ks[cur ^ 1][r][c], krow + (size_t)(kb + 32 + r) * HEADD + c);
        int d = idx >> 2, c2 = (idx & 3) * 8;
        cp16(&Vs[cur ^ 1][d][c2], vrow + (size_t)d * SEQ + kb + 32 + c2);
      }
    }

    // partial scores (K-range [wd*128, wd*128+128)), interleaved 2-deep
    v8f s0 = {}, s1 = {};
    const bf16* qt  = &Qs[wq * 16 + ln][0];
    const bf16* kt0 = &Ks[cur][ln][0];
    const bf16* kt1 = &Ks[cur][16 + ln][0];
    {
      const int ea = (wd * 4) * 32, eb = (wd * 4 + 1) * 32;
      v16bf aqA = ld_frag(qt,  ea + half * 16, ea + half * 16 + 8);
      v16bf b0A = ld_frag(kt0, ea + half * 16, ea + half * 16 + 8);
      v16bf b1A = ld_frag(kt1, ea + half * 16, ea + half * 16 + 8);
      v16bf aqB = ld_frag(qt,  eb + half * 16, eb + half * 16 + 8);
      v16bf b0B = ld_frag(kt0, eb + half * 16, eb + half * 16 + 8);
      v16bf b1B = ld_frag(kt1, eb + half * 16, eb + half * 16 + 8);
#pragma unroll
      for (int j = 0; j < 4; j += 2) {
        s0 = wmma_bf16(aqA, b0A, s0);
        s1 = wmma_bf16(aqA, b1A, s1);
        if (j + 2 < 4) {
          const int en = (wd * 4 + j + 2) * 32;
          aqA = ld_frag(qt,  en + half * 16, en + half * 16 + 8);
          b0A = ld_frag(kt0, en + half * 16, en + half * 16 + 8);
          b1A = ld_frag(kt1, en + half * 16, en + half * 16 + 8);
        }
        s0 = wmma_bf16(aqB, b0B, s0);
        s1 = wmma_bf16(aqB, b1B, s1);
        if (j + 3 < 4) {
          const int en = (wd * 4 + j + 3) * 32;
          aqB = ld_frag(qt,  en + half * 16, en + half * 16 + 8);
          b0B = ld_frag(kt0, en + half * 16, en + half * 16 + 8);
          b1B = ld_frag(kt1, en + half * 16, en + half * 16 + 8);
        }
      }
    }
#pragma unroll
    for (int r = 0; r < 8; ++r) {
      Sp[wave][0][half * 8 + r][ln] = s0[r];
      Sp[wave][1][half * 8 + r][ln] = s1[r];
    }
    __syncthreads();
    // combine the 4 partials of this query slice (replicated, cheap VALU)
#pragma unroll
    for (int r = 0; r < 8; ++r) {
      float t0 = 0.0f, t1 = 0.0f;
#pragma unroll
      for (int w = 0; w < 4; ++w) {
        t0 += Sp[wq * 4 + w][0][half * 8 + r][ln];
        t1 += Sp[wq * 4 + w][1][half * 8 + r][ln];
      }
      s0[r] = t0; s1[r] = t1;
    }

    // scale *64, additive key mask (col = lane)
    const int m0 = Ms[kb + ln];
    const int m1 = Ms[kb + 16 + ln];
    float t[8], corr[8];
#pragma unroll
    for (int r = 0; r < 8; ++r) {
      float a0 = m0 ? s0[r] * SCALE_F : NEG_F;
      float a1 = m1 ? s1[r] * SCALE_F : NEG_F;
      s0[r] = a0; s1[r] = a1;
      t[r] = fmaxf(a0, a1);
    }
#pragma unroll
    for (int off = 8; off > 0; off >>= 1)
#pragma unroll
      for (int r = 0; r < 8; ++r) t[r] = fmaxf(t[r], __shfl_xor(t[r], off, 32));
#pragma unroll
    for (int r = 0; r < 8; ++r) {                // online-softmax update
      float nm = fmaxf(mx[r], t[r]);
      corr[r] = __expf(mx[r] - nm);
      mx[r] = nm;
      s0[r] = __expf(s0[r] - nm);
      s1[r] = __expf(s1[r] - nm);
      t[r] = s0[r] + s1[r];
    }
#pragma unroll
    for (int off = 8; off > 0; off >>= 1)
#pragma unroll
      for (int r = 0; r < 8; ++r) t[r] += __shfl_xor(t[r], off, 32);
#pragma unroll
    for (int r = 0; r < 8; ++r) sm[r] = sm[r] * corr[r] + t[r];
#pragma unroll
    for (int f = 0; f < 8; ++f)                  // rescale running output
#pragma unroll
      for (int r = 0; r < 8; ++r) acc[f][r] *= corr[r];

    // Relayout P (D layout -> A layout) through the per-wave LDS buffer.
#pragma unroll
    for (int r = 0; r < 8; ++r) {
      Ps[wave][half * 8 + r][ln]      = (bf16)s0[r];
      Ps[wave][half * 8 + r][16 + ln] = (bf16)s1[r];
    }
    v16bf pa = ld_frag(&Ps[wave][ln][0], half * 16, half * 16 + 8);

    // out += P(16x32) @ V(32x128), interleaved 2-deep V pipeline
    {
      const bf16* vr = &Vs[cur][wd * 128 + ln][0];
      v16bf vA = ld_frag(vr,           half * 16, half * 16 + 8);  // f=0
      v16bf vB = ld_frag(vr + 16 * 40, half * 16, half * 16 + 8);  // f=1
#pragma unroll
      for (int f = 0; f < 8; f += 2) {
        acc[f] = wmma_bf16(pa, vA, acc[f]);
        if (f + 2 < 8)
          vA = ld_frag(vr + (f + 2) * 16 * 40, half * 16, half * 16 + 8);
        acc[f + 1] = wmma_bf16(pa, vB, acc[f + 1]);
        if (f + 3 < 8)
          vB = ld_frag(vr + (f + 3) * 16 * 40, half * 16, half * 16 + 8);
      }
    }
    async_wait_all();                            // next tile landed in LDS
    __syncthreads();                             // all reads of cur done
    cur ^= 1;
  }

  // Normalize and store fp32 output [B,S,512]
  float* __restrict__ orow = out + ((size_t)b * SEQ + qbase) * HEADD;
#pragma unroll
  for (int f = 0; f < 8; ++f) {
    int col = wd * 128 + f * 16 + ln;
#pragma unroll
    for (int r = 0; r < 8; ++r) {
      int qr = wq * 16 + half * 8 + r;
      orow[(size_t)qr * HEADD + col] = acc[f][r] / sm[r];
    }
  }
}

// ---------------------------------------------------------------------------
extern "C" void kernel_launch(void* const* d_in, const int* in_sizes, int n_in,
                              void* d_out, int out_size, void* d_ws, size_t ws_size,
                              hipStream_t stream) {
  (void)in_sizes; (void)n_in; (void)out_size; (void)ws_size;
  const float* q    = (const float*)d_in[0];
  const float* k    = (const float*)d_in[1];
  const float* v    = (const float*)d_in[2];
  const int*   mask = (const int*)d_in[3];
  const float* Wq   = (const float*)d_in[4];
  const float* Wk   = (const float*)d_in[5];
  const float* Wv   = (const float*)d_in[6];
  float* out = (float*)d_out;

  // workspace: qh | kh | vt   (each 8192*512 bf16 = 8 MiB, total 24 MiB)
  bf16* qh = (bf16*)d_ws;
  bf16* kh = qh + (size_t)ROWS * HEADD;
  bf16* vt = kh + (size_t)ROWS * HEADD;

  dim3 gProj(ROWS / 128, 3);
  qkv_proj_kernel<<<gProj, 512, 0, stream>>>(q, k, v, Wq, Wk, Wv, qh, kh, vt);

  dim3 gAttn(SEQ / 32, NBATCH);
  flash_attn_kernel<<<gAttn, 256, 0, stream>>>(qh, kh, vt, mask, out);
}
